// LAVAMemory_80685255622735
// MI455X (gfx1250) — matrix operand order repaired and checked by reading, not compiled, not run on hardware
//
#include <hip/hip_runtime.h>

// ---------------------------------------------------------------------------
// Problem constants (match reference setup_inputs)
// ---------------------------------------------------------------------------
#define TOKENS   2048      // B*S = 2*1024
#define HDIM     1024
#define LSLOTS   16384
#define NCLUST   128
#define SPC      128       // slots per cluster
#define TOPK     8

typedef __attribute__((ext_vector_type(16))) __bf16 v16bf;
typedef __attribute__((ext_vector_type(8)))  float  v8f;

// ---------------------------------------------------------------------------
// bf16 <-> f32 bit helpers (we keep bf16 as raw ushort everywhere except WMMA)
// ---------------------------------------------------------------------------
__device__ __forceinline__ float bf2f(unsigned short u) {
    union { unsigned int i; float f; } c; c.i = ((unsigned int)u) << 16; return c.f;
}
__device__ __forceinline__ unsigned short f2bf(float f) {
    union { float f; unsigned int i; } c; c.f = f;
    unsigned int r = c.i + 0x7FFFu + ((c.i >> 16) & 1u);   // round-to-nearest-even
    return (unsigned short)(r >> 16);
}
union PK4 { unsigned short s[4]; uint2 u; };

// ---------------------------------------------------------------------------
// Kernel 1: fp32 -> bf16 (vectorized, 4 elems/thread)
// ---------------------------------------------------------------------------
__global__ void f32_to_bf16_kernel(const float* __restrict__ src,
                                   unsigned short* __restrict__ dst, int n4) {
    int i = blockIdx.x * blockDim.x + threadIdx.x;
    if (i >= n4) return;
    float4 v = ((const float4*)src)[i];
    PK4 p;
    p.s[0] = f2bf(v.x); p.s[1] = f2bf(v.y); p.s[2] = f2bf(v.z); p.s[3] = f2bf(v.w);
    ((uint2*)dst)[i] = p.u;
}

// ---------------------------------------------------------------------------
// Kernel 2: int32 -> int8 repack of contents (64MB -> 16MB, L2 resident after)
// ---------------------------------------------------------------------------
__global__ void pack_i8_kernel(const int* __restrict__ src,
                               signed char* __restrict__ dst, int n4) {
    int i = blockIdx.x * blockDim.x + threadIdx.x;
    if (i >= n4) return;
    int4 v = ((const int4*)src)[i];
    char4 c;
    c.x = (signed char)v.x; c.y = (signed char)v.y;
    c.z = (signed char)v.z; c.w = (signed char)v.w;
    ((char4*)dst)[i] = c;
}

// ---------------------------------------------------------------------------
// Kernel 3: per-slot L2-normalized address rows -> bf16  (wave per slot)
// ---------------------------------------------------------------------------
__global__ __launch_bounds__(256)
void addr_norm_kernel(const float* __restrict__ addr,
                      unsigned short* __restrict__ out) {
    const int lane = threadIdx.x & 31, wave = threadIdx.x >> 5;
    const int slot = blockIdx.x * 8 + wave;
    const float* row = addr + (size_t)slot * HDIM;
    float4 v[8];
    float ss = 0.f;
#pragma unroll
    for (int j = 0; j < 8; ++j) {
        v[j] = ((const float4*)row)[j * 32 + lane];
        ss += v[j].x * v[j].x + v[j].y * v[j].y + v[j].z * v[j].z + v[j].w * v[j].w;
    }
#pragma unroll
    for (int o = 16; o; o >>= 1) ss += __shfl_xor(ss, o, 32);
    const float inv = 1.f / fmaxf(sqrtf(ss), 1e-8f);
    uint2* orow = (uint2*)(out + (size_t)slot * HDIM);
#pragma unroll
    for (int j = 0; j < 8; ++j) {
        PK4 p;
        p.s[0] = f2bf(v[j].x * inv); p.s[1] = f2bf(v[j].y * inv);
        p.s[2] = f2bf(v[j].z * inv); p.s[3] = f2bf(v[j].w * inv);
        orow[j * 32 + lane] = p.u;
    }
}

// ---------------------------------------------------------------------------
// Kernel 4: per-cluster centroid (mean over 128 round-robin slots), normalize
// ---------------------------------------------------------------------------
__global__ __launch_bounds__(256)
void centroid_kernel(const float* __restrict__ addr,
                     unsigned short* __restrict__ cent) {
    const int c = blockIdx.x, t = threadIdx.x;
    const int lane = t & 31, wave = t >> 5;
    float4 s = {0.f, 0.f, 0.f, 0.f};
    for (int i = 0; i < SPC; ++i) {
        float4 v = ((const float4*)(addr + (size_t)(i * NCLUST + c) * HDIM))[t];
        s.x += v.x; s.y += v.y; s.z += v.z; s.w += v.w;
    }
    const float inv_spc = 1.f / (float)SPC;
    s.x *= inv_spc; s.y *= inv_spc; s.z *= inv_spc; s.w *= inv_spc;
    float ss = s.x * s.x + s.y * s.y + s.z * s.z + s.w * s.w;
#pragma unroll
    for (int o = 16; o; o >>= 1) ss += __shfl_xor(ss, o, 32);
    __shared__ float red[8];
    __shared__ float sinv;
    if (lane == 0) red[wave] = ss;
    __syncthreads();
    if (t == 0) {
        float tot = 0.f;
        for (int w = 0; w < 8; ++w) tot += red[w];
        sinv = 1.f / fmaxf(sqrtf(tot), 1e-8f);
    }
    __syncthreads();
    PK4 p;
    p.s[0] = f2bf(s.x * sinv); p.s[1] = f2bf(s.y * sinv);
    p.s[2] = f2bf(s.z * sinv); p.s[3] = f2bf(s.w * sinv);
    ((uint2*)(cent + (size_t)c * HDIM))[t] = p.u;
}

// ---------------------------------------------------------------------------
// WMMA GEMM:  C[M,N] (f32) = A[M,K](bf16) * W[N,K](bf16)^T
// 128x128 block tile, K-step 32, 8 waves (4M x 2N), each wave: 2x4 16x16 frags.
// Double-buffered LDS fed with global_load_async_to_lds_b128 (ASYNCcnt).
// Per-thread global pointers are precomputed once and bumped by TK*2 bytes
// per fill, so each fill is just 4 async issues + 4 pointer adds.
// ---------------------------------------------------------------------------
#define TM 128
#define TN 128
#define TK 32

__device__ __forceinline__ void async_ld16(unsigned lds_off, const unsigned short* gp) {
    asm volatile("global_load_async_to_lds_b128 %0, %1, off"
                 :: "v"(lds_off), "v"((unsigned long long)(size_t)gp)
                 : "memory");
}

union ABfrag { v16bf v; uint4 u[2]; };

__global__ __launch_bounds__(256)
void gemm_bf16_wmma(const unsigned short* __restrict__ A,
                    const unsigned short* __restrict__ Bw,
                    float* __restrict__ C, int M, int N, int K) {
    __shared__ __align__(16) unsigned short As[2][TM * TK];
    __shared__ __align__(16) unsigned short Bs[2][TN * TK];

    const int tid  = threadIdx.x;
    const int lane = tid & 31, wave = tid >> 5;
    const int bm = blockIdx.y * TM;
    const int bn = blockIdx.x * TN;
    const int mbase = (wave >> 1) * 32;   // wave grid 4(M) x 2(N)
    const int nbase = (wave & 1) * 64;
    const int hi = lane >> 4;             // lane half-select
    const int lm = lane & 15;

    // ---- precomputed per-thread load slots (2 A + 2 B per thread per fill)
    const int u0 = tid, u1 = tid + 256;
    const int r0 = u0 >> 2, g0 = u0 & 3;   // row 0..63,  8-half group
    const int r1 = u1 >> 2, g1 = u1 & 3;   // row 64..127
    const unsigned aOff0 = (unsigned)((r0 * TK + g0 * 8) * 2);
    const unsigned aOff1 = (unsigned)((r1 * TK + g1 * 8) * 2);
    const unsigned asBase[2] = { (unsigned)(size_t)&As[0][0], (unsigned)(size_t)&As[1][0] };
    const unsigned bsBase[2] = { (unsigned)(size_t)&Bs[0][0], (unsigned)(size_t)&Bs[1][0] };
    const unsigned short* gA0 = A  + (size_t)(bm + r0) * K + g0 * 8;
    const unsigned short* gA1 = A  + (size_t)(bm + r1) * K + g1 * 8;
    const unsigned short* gB0 = Bw + (size_t)(bn + r0) * K + g0 * 8;
    const unsigned short* gB1 = Bw + (size_t)(bn + r1) * K + g1 * 8;

    v8f acc[2][4] = {};
    const int KT = K / TK;

    // prologue: fill buffer 0
    async_ld16(asBase[0] + aOff0, gA0);
    async_ld16(asBase[0] + aOff1, gA1);
    async_ld16(bsBase[0] + aOff0, gB0);
    async_ld16(bsBase[0] + aOff1, gB1);
    gA0 += TK; gA1 += TK; gB0 += TK; gB1 += TK;

    int buf = 0;
    for (int kt = 0; kt < KT; ++kt) {
        if (kt + 1 < KT) {
            const int nb = buf ^ 1;
            async_ld16(asBase[nb] + aOff0, gA0);
            async_ld16(asBase[nb] + aOff1, gA1);
            async_ld16(bsBase[nb] + aOff0, gB0);
            async_ld16(bsBase[nb] + aOff1, gB1);
            gA0 += TK; gA1 += TK; gB0 += TK; gB1 += TK;
            asm volatile("s_wait_asynccnt 0x4" ::: "memory");
        } else {
            asm volatile("s_wait_asynccnt 0x0" ::: "memory");
        }
        __syncthreads();

        // A fragments (16x32 bf16): lanes 0-15 rows, halves K=0..7/16..23 (hi:+8)
        ABfrag a[2];
#pragma unroll
        for (int mf = 0; mf < 2; ++mf) {
            const uint4* ap = (const uint4*)&As[buf][(mbase + mf * 16 + lm) * TK];
            a[mf].u[0] = ap[hi];
            a[mf].u[1] = ap[hi + 2];
        }
#pragma unroll
        for (int nf = 0; nf < 4; ++nf) {
            // B fragment (32x16): lane = col, halves K=0..7/16..23 (hi:+8)
            const uint4* bp = (const uint4*)&Bs[buf][(nbase + nf * 16 + lm) * TK];
            ABfrag b;
            b.u[0] = bp[hi];
            b.u[1] = bp[hi + 2];
#pragma unroll
            for (int mf = 0; mf < 2; ++mf) {
                acc[mf][nf] = __builtin_amdgcn_wmma_f32_16x16x32_bf16(
                    false, a[mf].v, false, b.v, (short)0, acc[mf][nf], false, false);
            }
        }
        __syncthreads();
        buf ^= 1;
    }

    // store: VGPR r -> row (r + 8*hi), col = lane&15
#pragma unroll
    for (int mf = 0; mf < 2; ++mf) {
#pragma unroll
        for (int nf = 0; nf < 4; ++nf) {
            const int row0 = bm + mbase + mf * 16 + hi * 8;
            const int col  = bn + nbase + nf * 16 + lm;
            float* cp = C + (size_t)row0 * N + col;
#pragma unroll
            for (int r = 0; r < 8; ++r) cp[(size_t)r * N] = acc[mf][nf][r];
        }
    }
}

// ---------------------------------------------------------------------------
// Kernel 6: per-token cluster argmax -> 128 candidate scores -> top-8 ->
//           softmax -> int8 dequant weighted combine -> bf16 read row.
// One block (8 waves) per token. All gather state is L2-resident.
// ---------------------------------------------------------------------------
__device__ __forceinline__ float dot_row_bf16(const float* qs,
                                              const unsigned short* row, int lane) {
    float d = 0.f;
#pragma unroll
    for (int j = 0; j < 8; ++j) {
        const int v4 = j * 32 + lane;             // float4 index (16B aligned)
        float4 qv = ((const float4*)qs)[v4];      // one ds_load_b128
        uint2 u = *(const uint2*)(row + v4 * 4);  // 4 bf16 from global/L2
        d = fmaf(qv.x, bf2f((unsigned short)(u.x & 0xffffu)), d);
        d = fmaf(qv.y, bf2f((unsigned short)(u.x >> 16)), d);
        d = fmaf(qv.z, bf2f((unsigned short)(u.y & 0xffffu)), d);
        d = fmaf(qv.w, bf2f((unsigned short)(u.y >> 16)), d);
    }
#pragma unroll
    for (int o = 16; o; o >>= 1) d += __shfl_xor(d, o, 32);
    return d;
}

__global__ __launch_bounds__(256)
void select_read_kernel(const float* __restrict__ q,
                        const unsigned short* __restrict__ cent,
                        const unsigned short* __restrict__ anorm,
                        const signed char* __restrict__ ci8,
                        const float* __restrict__ cscale,
                        unsigned short* __restrict__ readout) {
    __shared__ __align__(16) float qs[HDIM];
    __shared__ float sc[SPC];
    __shared__ float bsv[8];
    __shared__ int   bsi[8];
    __shared__ float topv[TOPK];
    __shared__ int   topi[TOPK];
    __shared__ float wts[TOPK];
    __shared__ int   slots[TOPK];
    __shared__ float qss_s, qinv_s;
    __shared__ int   cbest_s;

    const int tid = threadIdx.x, lane = tid & 31, wave = tid >> 5;
    const int tok = blockIdx.x;

    // load q row + ||q||^2
    float4 qv = ((const float4*)(q + (size_t)tok * HDIM))[tid];
    ((float4*)qs)[tid] = qv;
    if (tid == 0) qss_s = 0.f;
    float ss = qv.x * qv.x + qv.y * qv.y + qv.z * qv.z + qv.w * qv.w;
#pragma unroll
    for (int o = 16; o; o >>= 1) ss += __shfl_xor(ss, o, 32);
    __syncthreads();
    if (lane == 0) atomicAdd(&qss_s, ss);
    __syncthreads();
    if (tid == 0) qinv_s = 1.f / fmaxf(sqrtf(qss_s), 1e-8f);

    // stage 1: argmax over 128 normalized centroids (scale-invariant for argmax)
    float bestv = -3.4e38f; int besti = 0;
    for (int ii = 0; ii < 16; ++ii) {
        int cc = wave * 16 + ii;
        float d = dot_row_bf16(qs, cent + (size_t)cc * HDIM, lane);
        if (d > bestv) { bestv = d; besti = cc; }
    }
    if (lane == 0) { bsv[wave] = bestv; bsi[wave] = besti; }
    __syncthreads();
    if (tid == 0) {
        float bv = bsv[0]; int bi = bsi[0];
        for (int w = 1; w < 8; ++w)
            if (bsv[w] > bv) { bv = bsv[w]; bi = bsi[w]; }
        cbest_s = bi;
    }
    __syncthreads();
    const int c = cbest_s;

    // stage 2: 128 candidate cosine scores (slot = i*128 + c)
    for (int ii = 0; ii < 16; ++ii) {
        int i = wave + ii * 8;
        float d = dot_row_bf16(qs, anorm + (size_t)(i * NCLUST + c) * HDIM, lane);
        if (lane == 0) sc[i] = d;
    }
    __syncthreads();

    // top-8 (tie-break lowest index, like lax.top_k) + softmax, by wave 0
    if (wave == 0) {
        for (int t8 = 0; t8 < TOPK; ++t8) {
            float v = -3.4e38f; int idx = 0;
#pragma unroll
            for (int b = 0; b < 4; ++b) {
                int j = b * 32 + lane;
                float s = sc[j];
                if (s > v || (s == v && j < idx)) { v = s; idx = j; }
            }
#pragma unroll
            for (int o = 16; o; o >>= 1) {
                float ov = __shfl_xor(v, o, 32);
                int   oi = __shfl_xor(idx, o, 32);
                if (ov > v || (ov == v && oi < idx)) { v = ov; idx = oi; }
            }
            if (lane == 0) { topv[t8] = v; topi[t8] = idx; sc[idx] = -3.4e38f; }
        }
        if (lane == 0) {
            const float qi = qinv_s;
            float m = -3.4e38f;
            for (int k = 0; k < TOPK; ++k) m = fmaxf(m, topv[k] * qi);
            float e[TOPK]; float sum = 0.f;
            for (int k = 0; k < TOPK; ++k) { e[k] = __expf(topv[k] * qi - m); sum += e[k]; }
            const float inv = 1.f / sum;
            for (int k = 0; k < TOPK; ++k) {
                wts[k]   = e[k] * inv;
                slots[k] = topi[k] * NCLUST + c;
            }
        }
    }
    __syncthreads();

    // combine: read = sum_k w_k * scale[slot_k] * int8_row  -> bf16
    float4 acc = {0.f, 0.f, 0.f, 0.f};
    const int d0 = tid * 4;
#pragma unroll
    for (int k = 0; k < TOPK; ++k) {
        const int slot = slots[k];
        const float wk = wts[k] * cscale[slot];
        char4 cv = *(const char4*)(ci8 + (size_t)slot * HDIM + d0);
        acc.x = fmaf(wk, (float)cv.x, acc.x);
        acc.y = fmaf(wk, (float)cv.y, acc.y);
        acc.z = fmaf(wk, (float)cv.z, acc.z);
        acc.w = fmaf(wk, (float)cv.w, acc.w);
    }
    PK4 p;
    p.s[0] = f2bf(acc.x); p.s[1] = f2bf(acc.y);
    p.s[2] = f2bf(acc.z); p.s[3] = f2bf(acc.w);
    ((uint2*)(readout + (size_t)tok * HDIM))[tid] = p.u;
}

// ---------------------------------------------------------------------------
// Host-side orchestration
// ---------------------------------------------------------------------------
extern "C" void kernel_launch(void* const* d_in, const int* in_sizes, int n_in,
                              void* d_out, int out_size, void* d_ws, size_t ws_size,
                              hipStream_t stream) {
    (void)in_sizes; (void)n_in; (void)out_size; (void)ws_size;

    const float* x          = (const float*)d_in[0];   // [2,1024,1024]
    const float* W_addr     = (const float*)d_in[1];   // [1024,1024]
    const float* W_read     = (const float*)d_in[2];   // [1024,1024]
    const float* addresses  = (const float*)d_in[3];   // [16384,1024]
    const int*   contents_q = (const int*)  d_in[4];   // [16384,1024]
    const float* cscale     = (const float*)d_in[5];   // [16384]
    float*       out        = (float*)d_out;           // [2,1024,1024]

    char* ws = (char*)d_ws;
    size_t off = 0;
    auto alloc = [&](size_t bytes) { char* p = ws + off; off = (off + bytes + 255) & ~(size_t)255; return p; };

    unsigned short* xbf    = (unsigned short*)alloc((size_t)TOKENS * HDIM * 2);  //  4 MB
    unsigned short* wabf   = (unsigned short*)alloc((size_t)HDIM * HDIM * 2);    //  2 MB
    unsigned short* wrbf   = (unsigned short*)alloc((size_t)HDIM * HDIM * 2);    //  2 MB
    unsigned short* anorm  = (unsigned short*)alloc((size_t)LSLOTS * HDIM * 2);  // 32 MB
    unsigned short* cent   = (unsigned short*)alloc((size_t)NCLUST * HDIM * 2);  // 256 KB
    signed char*    ci8    = (signed char*)  alloc((size_t)LSLOTS * HDIM);       // 16 MB
    float*          qf32   = (float*)        alloc((size_t)TOKENS * HDIM * 4);   //  8 MB
    unsigned short* readbf = (unsigned short*)alloc((size_t)TOKENS * HDIM * 2);  //  4 MB

    // 1) precision conversion + repack (one streaming pass over the big inputs)
    f32_to_bf16_kernel<<<(TOKENS * HDIM / 4) / 256, 256, 0, stream>>>(x, xbf, TOKENS * HDIM / 4);
    f32_to_bf16_kernel<<<(HDIM * HDIM / 4) / 256, 256, 0, stream>>>(W_addr, wabf, HDIM * HDIM / 4);
    f32_to_bf16_kernel<<<(HDIM * HDIM / 4) / 256, 256, 0, stream>>>(W_read, wrbf, HDIM * HDIM / 4);
    pack_i8_kernel<<<(LSLOTS * HDIM / 4) / 256, 256, 0, stream>>>(contents_q, ci8, LSLOTS * HDIM / 4);
    addr_norm_kernel<<<LSLOTS / 8, 256, 0, stream>>>(addresses, anorm);
    centroid_kernel<<<NCLUST, 256, 0, stream>>>(addresses, cent);

    // 2) q = x @ W_addr^T   (WMMA bf16, fp32 accum)
    gemm_bf16_wmma<<<dim3(HDIM / TN, TOKENS / TM), 256, 0, stream>>>(
        xbf, wabf, qf32, TOKENS, HDIM, HDIM);

    // 3) IVF select + top-8 + softmax + int8 dequant combine (L2-resident state)
    select_read_kernel<<<TOKENS, 256, 0, stream>>>(qf32, cent, anorm, ci8, cscale, readbf);

    // 4) out = read @ W_read^T  (WMMA bf16, fp32 out)
    gemm_bf16_wmma<<<dim3(HDIM / TN, TOKENS / TM), 256, 0, stream>>>(
        readbf, wrbf, out, TOKENS, HDIM, HDIM);
}
